// CorrBlock_9878424780867
// MI455X (gfx1250) — compile-verified, matrix operand order checked by reading, not run on hardware
//
#include <hip/hip_runtime.h>
#include <hip/hip_bf16.h>
#include <stdint.h>
#include <string.h>

// ---- CDNA5 vector types (probe-confirmed builtin signatures) ----
typedef __attribute__((ext_vector_type(16))) __bf16        v16bf;
typedef __attribute__((ext_vector_type(8)))  float         v8f;
typedef __attribute__((ext_vector_type(4)))  unsigned int  u32x4;
typedef __attribute__((ext_vector_type(8)))  int           i32x8;
typedef __attribute__((ext_vector_type(4)))  int           i32x4;

// Problem constants (fixed by the reference)
#define CN   8
#define CC   256
#define CH   80
#define CW   160
#define CHW  (CH * CW)          // 12800
#define NDISP 81                // 9x9 window
#define TX   16                 // x-tile per workgroup

#define AROW 18                 // dwords per channel row of LDS A (16 px + 2 pad)
#define BROW 18                 // uint32 (bf16-pairs) per u row of LDS B (16 pairs + 2 pad)

// float pair -> packed bf16 pair (lo in [15:0]).  The clang builtin gets
// scalarized into cvt+shift+merge, so on device we emit the packed convert
// directly (mnemonic verified in this toolchain's own gfx1250 output).
__device__ __forceinline__ uint32_t pack_bf16_pair(float a, float b) {
#if defined(__gfx1250__)
    uint32_t r;
    asm("v_cvt_pk_bf16_f32 %0, %1, %2" : "=v"(r) : "v"(a), "v"(b));
    return r;
#else
    // Host pass / generic fallback: integer RNE.
    uint32_t ua = __float_as_uint(a);
    uint32_t ub = __float_as_uint(b);
    uint32_t ra = (ua + 0x7FFFu + ((ua >> 16) & 1u)) >> 16;
    uint32_t rb = (ub + 0x7FFFu + ((ub >> 16) & 1u)) >> 16;
    return (ra & 0xFFFFu) | (rb << 16);
#endif
}

__global__ __launch_bounds__(32)
void corr81_wmma_bf16(const float* __restrict__ f1,
                      const float* __restrict__ f2,
                      float* __restrict__ out)
{
    // LDS: A tile (f32, rows = channel, 16 px + pad), double-buffered B chunk
    // (bf16 pairs, rows = u, 32 ch + pad), output staging tile [81 x 16].
    __shared__ uint32_t ldsA[CC * AROW];             // 18432 B
    __shared__ uint32_t ldsB[2][32 * BROW];          //  4608 B
    __shared__ __align__(16) float ldsO[NDISP * TX]; //  5184 B

    const int lane = threadIdx.x;             // 0..31 (wave32)
    const int half = lane >> 4;               // 0 / 1
    const int nn   = lane & 15;

    const int tile = blockIdx.x;              // N*H*(W/16) = 6400 tiles
    const int xt = tile % (CW / TX);
    const int y  = (tile / (CW / TX)) % CH;
    const int n  =  tile / ((CW / TX) * CH);
    const int x0 = xt * TX;

    const float* f1n = f1 + (size_t)n * CC * CHW;
    const float* f2n = f2 + (size_t)n * CC * CHW;

    // ---------------------------------------------------------------
    // Stage A = feat1[n, 0:256, y, x0:x0+16] into LDS via the TDM.
    // 2-D descriptor: dim0 = x (16 f32, contiguous), dim1 = channel
    // (stride H*W), LDS pad of 2 dwords every 16 dwords -> 18-dword rows.
    // ---------------------------------------------------------------
#if defined(__gfx1250__) && __has_builtin(__builtin_amdgcn_tensor_load_to_lds)
    {
        const uint64_t ga = (uint64_t)(uintptr_t)(f1n + (size_t)y * CW + x0);
        const uint32_t la = (uint32_t)(uintptr_t)(&ldsA[0]);  // LDS offset (flat low 32)
        u32x4 g0;
        g0[0] = 1u;                                            // count=1, user D#
        g0[1] = la;                                            // lds_addr
        g0[2] = (uint32_t)ga;                                  // global_addr[31:0]
        g0[3] = (uint32_t)((ga >> 32) & 0x01FFFFFFu)           // global_addr[56:32]
              | (2u << 30);                                    // type = 2 ("image")
        i32x8 g1;
        g1[0] = (int)((2u << 16)      // data_size = 4 B
                    | (1u << 20)      // pad_enable
                    | (3u << 22)      // pad_interval: 16 dwords
                    | (1u << 25));    // pad_amount:   2 dwords
        g1[1] = (int)(16u  << 16);    // tensor_dim0 = 16  (bits 79:48)
        g1[2] = (int)(256u << 16);    // dim0 hi = 0 | tensor_dim1 = 256 (bits 111:80)
        g1[3] = (int)(16u  << 16);    // dim1 hi = 0 | tile_dim0 = 16    (bits 127:112)
        g1[4] = (int)(256u);          // tile_dim1 = 256, tile_dim2 = 0
        g1[5] = (int)(CHW);           // tensor_dim0_stride = 12800 (48-bit lo)
        g1[6] = 0;                    // stride0 hi | stride1 lo
        g1[7] = 0;                    // stride1 hi
        i32x4 z4 = {0, 0, 0, 0};      // groups 2/3 unused (<=2D tensor)
        i32x8 z8 = {0, 0, 0, 0, 0, 0, 0, 0};
        __builtin_amdgcn_tensor_load_to_lds(g0, g1, z4, z4, z8, 0);
    }
  #if __has_builtin(__builtin_amdgcn_s_wait_tensorcnt)
    __builtin_amdgcn_s_wait_tensorcnt((short)0);
  #endif
    asm volatile("" ::: "memory");    // keep LDS reads after the TDM wait
    __syncthreads();
#else
    // Fallback (host pass / toolchains without the builtin): cooperative load.
    for (int i = lane; i < CC * TX; i += 32) {
        const int c = i >> 4, m = i & 15;
        ldsA[c * AROW + m] = __float_as_uint(f1n[(size_t)c * CHW + (size_t)y * CW + x0 + m]);
    }
    __syncthreads();
#endif

    // ---------------------------------------------------------------
    // Build the 8 A fragments (16x32 bf16 each) once; layout per ISA:
    // lane = {half, m}, VGPR v holds K pair 2*p, p = (v<4 ? v : 8+v-4) + 4*half.
    // ---------------------------------------------------------------
    v16bf afrag[8];
#pragma unroll
    for (int ks = 0; ks < 8; ++ks) {
        union { v16bf v; uint32_t u[8]; } af;
#pragma unroll
        for (int v = 0; v < 8; ++v) {
            const int p = ((v < 4) ? v : (8 + (v - 4))) + 4 * half;
            const int k = ks * 32 + 2 * p;
            const float lo = __uint_as_float(ldsA[(k)     * AROW + nn]);
            const float hi = __uint_as_float(ldsA[(k + 1) * AROW + nn]);
            af.u[v] = pack_bf16_pair(lo, hi);
        }
        afrag[ks] = af.v;
    }

    // ---------------------------------------------------------------
    // For each dy: accumulate O[16 x 32] = A(16x256) * B(256x32) over 8
    // K-steps of two 16x16x32 bf16 WMMAs, then scatter the 9-wide band.
    // B staging: lane = u column; x padding via clamped address +
    // branch-free cndmask on the packed pair (no EXEC branching).
    // ---------------------------------------------------------------
    const int xg   = x0 - 4 + lane;                       // this lane's u column
    const bool xin = (xg >= 0) && (xg < CW);
    const int xgc  = xg < 0 ? 0 : (xg >= CW ? CW - 1 : xg);

    for (int dy = 0; dy < 9; ++dy) {
        const int yp = y + dy - 4;                 // wave-uniform
        union { v8f v; float f[8]; } acc0, acc1;
#pragma unroll
        for (int r = 0; r < 8; ++r) { acc0.f[r] = 0.0f; acc1.f[r] = 0.0f; }

        if (yp >= 0 && yp < CH) {
            const float* bbase = f2n + (size_t)yp * CW + xgc;   // clamped, always valid
            for (int ks = 0; ks < 8; ++ks) {
                const int buf = ks & 1;
                // Stage 32ch x 32u chunk; paired ds_store_b64.
#pragma unroll
                for (int cp = 0; cp < 16; cp += 2) {
                    const int c = ks * 32 + 2 * cp;
                    const float a0 = bbase[(size_t)(c + 0) * CHW];
                    const float a1 = bbase[(size_t)(c + 1) * CHW];
                    const float a2 = bbase[(size_t)(c + 2) * CHW];
                    const float a3 = bbase[(size_t)(c + 3) * CHW];
                    uint32_t p0 = pack_bf16_pair(a0, a1);
                    uint32_t p1 = pack_bf16_pair(a2, a3);
                    p0 = xin ? p0 : 0u;            // single cndmask per pair
                    p1 = xin ? p1 : 0u;
                    uint2 st; st.x = p0; st.y = p1;
                    *(uint2*)&ldsB[buf][lane * BROW + cp] = st;
                }
                if (ks < 7)                         // uniform guard, no divergence
                    __builtin_prefetch(bbase + (size_t)(ks * 32 + 32) * CHW, 0, 1);
                __syncthreads();                    // store -> fragment-read

                // Two B fragments (32x16 bf16): lane = {half, n=u%16}.
                union { v16bf v; uint32_t u[8]; } bf0, bf1;
#pragma unroll
                for (int v = 0; v < 8; ++v) {
                    const int p = ((v < 4) ? v : (8 + (v - 4))) + 4 * half;
                    bf0.u[v] = ldsB[buf][(nn)      * BROW + p];
                    bf1.u[v] = ldsB[buf][(nn + 16) * BROW + p];
                }
                acc0.v = __builtin_amdgcn_wmma_f32_16x16x32_bf16(
                             false, afrag[ks], false, bf0.v,
                             (short)0, acc0.v, false, false);
                acc1.v = __builtin_amdgcn_wmma_f32_16x16x32_bf16(
                             false, afrag[ks], false, bf1.v,
                             (short)0, acc1.v, false, false);
                // No trailing barrier: next iteration writes the other buffer;
                // the WAR on this buffer is covered by the next barrier.
            }
        }

        // Band extraction: D layout => lane holds column n, rows m = r + 8*half.
        // Output (m, dx): u_chunk = m + dx, dx in [0,8].
#pragma unroll
        for (int r = 0; r < 8; ++r) {
            const int m = r + 8 * half;
            const int dx0 = nn - m;
            if (dx0 >= 0 && dx0 <= 8)
                ldsO[(dy * 9 + dx0) * TX + m] = acc0.f[r];
            const int dx1 = (nn + 16) - m;
            if (dx1 >= 0 && dx1 <= 8)
                ldsO[(dy * 9 + dx1) * TX + m] = acc1.f[r];
        }
    }
    __syncthreads();

    // Coalesced writeback: 1/256 scale + LeakyReLU(0.1) fused; float2 stores.
    for (int i = lane; i < NDISP * TX / 2; i += 32) {
        const int d = i >> 3, mp = (i & 7) * 2;
        float v0 = ldsO[d * TX + mp]     * (1.0f / 256.0f);
        float v1 = ldsO[d * TX + mp + 1] * (1.0f / 256.0f);
        v0 = (v0 > 0.0f) ? v0 : 0.1f * v0;
        v1 = (v1 > 0.0f) ? v1 : 0.1f * v1;
        float2 o; o.x = v0; o.y = v1;
        *(float2*)&out[(((size_t)n * NDISP + d) * CH + y) * CW + x0 + mp] = o;
    }
}

extern "C" void kernel_launch(void* const* d_in, const int* in_sizes, int n_in,
                              void* d_out, int out_size, void* d_ws, size_t ws_size,
                              hipStream_t stream) {
    const float* f1 = (const float*)d_in[0];
    const float* f2 = (const float*)d_in[1];
    float* out = (float*)d_out;
    const int ntiles = CN * CH * (CW / TX);   // 8*80*10 = 6400
    corr81_wmma_bf16<<<dim3(ntiles), dim3(32), 0, stream>>>(f1, f2, out);
}
